// Attention_78580721647617
// MI455X (gfx1250) — matrix operand order repaired and checked by reading, not compiled, not run on hardware
//
#include <hip/hip_runtime.h>
#include <hip/hip_bf16.h>
#include <stdint.h>

typedef __bf16 v16bf __attribute__((ext_vector_type(16)));
typedef float  v8f   __attribute__((ext_vector_type(8)));
typedef unsigned int u32x8 __attribute__((ext_vector_type(8)));

#define S_LEN  2048
#define DMODEL 1024
#define NHEADS 16
#define HDIM   64
#define BATCH  4

// f32 -> bf16 (round to nearest even) as raw bits
static __device__ __forceinline__ unsigned short f2bf_bits(float f) {
    unsigned u = __float_as_uint(f);
    u += 0x7FFFu + ((u >> 16) & 1u);
    return (unsigned short)(u >> 16);
}

// A-operand: build 16 bf16 from two non-adjacent 16-byte chunks.
static __device__ __forceinline__ v16bf make_op(const unsigned short* p0,
                                                const unsigned short* p1) {
    uint4 a = *(const uint4*)p0;
    uint4 b = *(const uint4*)p1;
    u32x8 t;
    t[0] = a.x; t[1] = a.y; t[2] = a.z; t[3] = a.w;
    t[4] = b.x; t[5] = b.y; t[6] = b.z; t[7] = b.w;
    return __builtin_bit_cast(v16bf, t);
}

// B-operand: 32 contiguous, 32B-aligned bytes -> single 8-dword vector load
// (guarantees adjacent VGPR allocation; no register-pairing copies).
static __device__ __forceinline__ v16bf load_op32(const unsigned short* p) {
    return __builtin_bit_cast(v16bf, *(const u32x8*)p);
}

// Async global -> LDS copy of 16 bytes (ASYNCcnt-tracked, CDNA5).
static __device__ __forceinline__ void async_ld16(unsigned lds_off,
                                                  const unsigned short* gptr) {
    asm volatile("global_load_async_to_lds_b128 %0, %1, off"
                 :: "v"(lds_off), "v"(gptr) : "memory");
}

// ---------------------------------------------------------------------------
// f32 -> bf16 bulk convert, 4 elements / thread
// ---------------------------------------------------------------------------
__global__ void cvt_f32_bf16_x4(const float* __restrict__ src,
                                unsigned short* __restrict__ dst, int n4) {
    int i = blockIdx.x * blockDim.x + threadIdx.x;
    if (i >= n4) return;
    float4 v = ((const float4*)src)[i];
    uint2 o;
    o.x = (unsigned)f2bf_bits(v.x) | ((unsigned)f2bf_bits(v.y) << 16);
    o.y = (unsigned)f2bf_bits(v.z) | ((unsigned)f2bf_bits(v.w) << 16);
    ((uint2*)dst)[i] = o;
}

// ---------------------------------------------------------------------------
// C = A(M x 1024) @ W(1024 x 1024)^T via bf16 WMMA, f32 accumulate.
// K loop unrolled x2 with ping-pong operand register sets: a set is reloaded
// (for k+64) immediately after its WMMAs consume it, so loads overlap the
// other set's WMMAs.
// mode 0: write bf16 [B,H,S,hd]   mode 1: bf16 [B,H,hd,S]   mode 2: f32 [M,N]
// ---------------------------------------------------------------------------
__global__ __launch_bounds__(128) void proj_gemm(
    const unsigned short* __restrict__ A,
    const unsigned short* __restrict__ W,
    void* __restrict__ out, int mode, float scale) {
    const int lane = threadIdx.x & 31;
    const int wave = threadIdx.x >> 5;
    const int lo = lane & 15;
    const int hi = lane >> 4;
    const int m0 = (blockIdx.x * 4 + wave) * 16;
    const int n0 = blockIdx.y * 64;

    const unsigned short* arow = A + (m0 + lo) * DMODEL;
    const unsigned short* wr[4];
#pragma unroll
    for (int t = 0; t < 4; ++t) wr[t] = W + (n0 + t * 16 + lo) * DMODEL;

    auto ldA = [&](int k) {
        return make_op(arow + k + hi * 8, arow + k + hi * 8 + 16);
    };
    auto ldB = [&](int t, int k) {
        return make_op(wr[t] + k + hi * 8, wr[t] + k + hi * 8 + 16);   // two 16B chunks, table k order
    };

    v8f acc[4] = {};
    v16bf a0, a1, b0[4], b1[4];
    a0 = ldA(0);
#pragma unroll
    for (int t = 0; t < 4; ++t) b0[t] = ldB(t, 0);
    a1 = ldA(32);
#pragma unroll
    for (int t = 0; t < 4; ++t) b1[t] = ldB(t, 32);

    for (int k = 0; k < DMODEL; k += 64) {
        const int k2 = (k + 64) & (DMODEL - 1);   // wraps on last iter (harmless)
        const int k3 = (k + 96) & (DMODEL - 1);
#pragma unroll
        for (int t = 0; t < 4; ++t)
            acc[t] = __builtin_amdgcn_wmma_f32_16x16x32_bf16(
                false, a0, false, b0[t], (short)0, acc[t], false, false);
        a0 = ldA(k2);
#pragma unroll
        for (int t = 0; t < 4; ++t) b0[t] = ldB(t, k2);
#pragma unroll
        for (int t = 0; t < 4; ++t)
            acc[t] = __builtin_amdgcn_wmma_f32_16x16x32_bf16(
                false, a1, false, b1[t], (short)0, acc[t], false, false);
        a1 = ldA(k3);
#pragma unroll
        for (int t = 0; t < 4; ++t) b1[t] = ldB(t, k3);
    }

#pragma unroll
    for (int t = 0; t < 4; ++t) {
#pragma unroll
        for (int j = 0; j < 8; ++j) {
            float val = acc[t][j] * scale;
            int m = m0 + j + hi * 8;       // C row (lane layout: +8 for hi half)
            int n = n0 + t * 16 + lo;      // C col
            if (mode == 2) {
                ((float*)out)[m * DMODEL + n] = val;
            } else {
                int b = m >> 11, s = m & (S_LEN - 1);
                int h = n >> 6,  d = n & (HDIM - 1);
                unsigned short bits = f2bf_bits(val);
                if (mode == 0)
                    ((unsigned short*)out)[((b * NHEADS + h) * S_LEN + s) * HDIM + d] = bits;
                else
                    ((unsigned short*)out)[((b * NHEADS + h) * HDIM + d) * S_LEN + s] = bits;
            }
        }
    }
}

// ---------------------------------------------------------------------------
// Flash attention. Block = 4 waves = 64 queries of one (b,h); wave = 16 rows.
// K/V tiles (32 keys) staged in double-buffered LDS via async global->LDS
// copies shared by all 4 waves. Scores computed transposed (K_tile @ Q^T) so
// the f32 score registers map directly onto the bf16 A-operand layout for P·V.
// ---------------------------------------------------------------------------
__global__ __launch_bounds__(128) void flash_attn(
    const unsigned short* __restrict__ Q,   // [B,H,S,hd] bf16, pre-scaled
    const unsigned short* __restrict__ K,   // [B,H,S,hd] bf16
    const unsigned short* __restrict__ Vt,  // [B,H,hd,S] bf16
    unsigned short* __restrict__ attn) {    // [B,S,D]    bf16
    __shared__ unsigned short Ks[2][32 * HDIM];   // 2 x 4 KB  [key][dim]
    __shared__ unsigned short Vs[2][HDIM * 32];   // 2 x 4 KB  [dim][key]

    const int tid  = threadIdx.x;
    const int lane = tid & 31;
    const int wave = tid >> 5;
    const int lo = lane & 15;
    const int hi = lane >> 4;
    const int bh = blockIdx.x;
    const int b = bh >> 4, h = bh & 15;
    const int q0 = blockIdx.y * 64 + wave * 16;

    const unsigned short* Qh = Q + bh * S_LEN * HDIM;
    const unsigned short* Kh = K + bh * S_LEN * HDIM;
    const unsigned short* Vh = Vt + bh * HDIM * S_LEN;

    // Q as B-operand (lane = query column), two K-dim chunks of 32
    const unsigned short* qrow = Qh + (q0 + lo) * HDIM;
    v16bf qop0 = make_op(qrow + hi * 8, qrow + hi * 8 + 16);
    v16bf qop1 = make_op(qrow + 32 + hi * 8, qrow + 32 + hi * 8 + 16);

    // async stage of one 32-key tile (4 KB K + 4 KB V^T), 4 x b128 per thread
    auto stage = [&](int bj, int kbn) {
        unsigned kbase = (unsigned)(uintptr_t)&Ks[bj][0];
        unsigned vbase = (unsigned)(uintptr_t)&Vs[bj][0];
        const unsigned short* Kg = Kh + kbn * HDIM;  // contiguous 4 KB
        const unsigned short* Vg = Vh + kbn;         // 64 rows, stride S
#pragma unroll
        for (int c = tid; c < 256; c += 128)
            async_ld16(kbase + c * 16, Kg + c * 8);
#pragma unroll
        for (int c = tid; c < 256; c += 128) {
            int row = c >> 2, sub = c & 3;
            async_ld16(vbase + c * 16, Vg + row * S_LEN + sub * 8);
        }
    };

    float m_i = -1e30f, l_i = 0.f;
    v8f o[4] = {};
    int buf = 0;
    stage(0, 0);

    for (int kb = 0; kb < S_LEN; kb += 32) {
        if (kb + 32 < S_LEN) {
            stage(buf ^ 1, kb + 32);
            asm volatile("s_wait_asynccnt 0x4" ::: "memory");  // current buf done
        } else {
            asm volatile("s_wait_asynccnt 0x0" ::: "memory");
        }
        __syncthreads();   // all waves' portions of current buf visible

        const unsigned short* ks = Ks[buf];
        const unsigned short* vs = Vs[buf];

        v8f s0 = {}, s1 = {};
        {
            const unsigned short* kr0 = ks + lo * HDIM;
            const unsigned short* kr1 = ks + (16 + lo) * HDIM;
            v16bf a00 = make_op(kr0 + hi * 8, kr0 + hi * 8 + 16);
            v16bf a01 = make_op(kr0 + 32 + hi * 8, kr0 + 32 + hi * 8 + 16);
            v16bf a10 = make_op(kr1 + hi * 8, kr1 + hi * 8 + 16);
            v16bf a11 = make_op(kr1 + 32 + hi * 8, kr1 + 32 + hi * 8 + 16);
            s0 = __builtin_amdgcn_wmma_f32_16x16x32_bf16(false, a00, false, qop0, (short)0, s0, false, false);
            s0 = __builtin_amdgcn_wmma_f32_16x16x32_bf16(false, a01, false, qop1, (short)0, s0, false, false);
            s1 = __builtin_amdgcn_wmma_f32_16x16x32_bf16(false, a10, false, qop0, (short)0, s1, false, false);
            s1 = __builtin_amdgcn_wmma_f32_16x16x32_bf16(false, a11, false, qop1, (short)0, s1, false, false);
        }

        // Online softmax: lane holds 16 keys of query `lo`; partner is lane^16.
        float mx = s0[0];
#pragma unroll
        for (int j = 1; j < 8; ++j) mx = fmaxf(mx, s0[j]);
#pragma unroll
        for (int j = 0; j < 8; ++j) mx = fmaxf(mx, s1[j]);
        mx = fmaxf(mx, __shfl_xor(mx, 16, 32));
        float m_new = fmaxf(m_i, mx);
        float alpha = __expf(m_i - m_new);
        float p0[8], p1[8], sum = 0.f;
#pragma unroll
        for (int j = 0; j < 8; ++j) { p0[j] = __expf(s0[j] - m_new); sum += p0[j]; }
#pragma unroll
        for (int j = 0; j < 8; ++j) { p1[j] = __expf(s1[j] - m_new); sum += p1[j]; }
        sum += __shfl_xor(sum, 16, 32);
        l_i = l_i * alpha + sum;
        m_i = m_new;

        // Pack P into the 16x32 bf16 A-operand (tile0 -> e0..7, tile1 -> e8..15)
        u32x8 pt;
#pragma unroll
        for (int e = 0; e < 4; ++e)
            pt[e] = (unsigned)f2bf_bits(p0[2 * e]) | ((unsigned)f2bf_bits(p0[2 * e + 1]) << 16);
#pragma unroll
        for (int e = 0; e < 4; ++e)
            pt[4 + e] = (unsigned)f2bf_bits(p1[2 * e]) | ((unsigned)f2bf_bits(p1[2 * e + 1]) << 16);
        v16bf pa = __builtin_bit_cast(v16bf, pt);

        // Rescale O: accumulator row j is query j+8*hi -> fetch its alpha
        float aj[8];
#pragma unroll
        for (int j = 0; j < 8; ++j) aj[j] = __shfl(alpha, j + hi * 8, 32);
#pragma unroll
        for (int t = 0; t < 4; ++t)
#pragma unroll
            for (int j = 0; j < 8; ++j) o[t][j] *= aj[j];

        // O += P @ V : B-operand from V^T tile in LDS (lane = dim column)
#pragma unroll
        for (int t = 0; t < 4; ++t) {
            v16bf bv = make_op(vs + (t * 16 + lo) * 32 + hi * 8, vs + (t * 16 + lo) * 32 + hi * 8 + 16);
            o[t] = __builtin_amdgcn_wmma_f32_16x16x32_bf16(
                false, pa, false, bv, (short)0, o[t], false, false);
        }

        __syncthreads();   // compute done before next iter overwrites buf^1
        buf ^= 1;
    }

    float rl = 1.0f / l_i;
    float rj[8];
#pragma unroll
    for (int j = 0; j < 8; ++j) rj[j] = __shfl(rl, j + hi * 8, 32);
#pragma unroll
    for (int t = 0; t < 4; ++t)
#pragma unroll
        for (int j = 0; j < 8; ++j) {
            float val = o[t][j] * rj[j];
            int q = q0 + j + hi * 8;
            int d = t * 16 + lo;
            attn[(b * S_LEN + q) * DMODEL + h * HDIM + d] = f2bf_bits(val);
        }
}

// ---------------------------------------------------------------------------
extern "C" void kernel_launch(void* const* d_in, const int* in_sizes, int n_in,
                              void* d_out, int out_size, void* d_ws, size_t ws_size,
                              hipStream_t stream) {
    const float* x  = (const float*)d_in[0];
    const float* Wq = (const float*)d_in[1];
    const float* Wk = (const float*)d_in[2];
    const float* Wv = (const float*)d_in[3];
    const float* Wo = (const float*)d_in[4];

    char* w = (char*)d_ws;
    const size_t MB = 1ull << 20;
    unsigned short* xb  = (unsigned short*)(w + 0 * MB);   // 16 MB  x bf16
    unsigned short* wqb = (unsigned short*)(w + 16 * MB);  // 2 MB
    unsigned short* wkb = (unsigned short*)(w + 18 * MB);
    unsigned short* wvb = (unsigned short*)(w + 20 * MB);
    unsigned short* wob = (unsigned short*)(w + 22 * MB);
    unsigned short* qb  = (unsigned short*)(w + 24 * MB);  // 16 MB [B,H,S,hd]
    unsigned short* kbf = (unsigned short*)(w + 40 * MB);  // 16 MB [B,H,S,hd]
    unsigned short* vtb = (unsigned short*)(w + 56 * MB);  // 16 MB [B,H,hd,S]
    unsigned short* ab  = (unsigned short*)(w + 72 * MB);  // 16 MB [B,S,D]

    int nx4 = (BATCH * S_LEN * DMODEL) / 4;
    cvt_f32_bf16_x4<<<(nx4 + 255) / 256, 256, 0, stream>>>(x, xb, nx4);
    int nw4 = (DMODEL * DMODEL) / 4;
    cvt_f32_bf16_x4<<<(nw4 + 255) / 256, 256, 0, stream>>>(Wq, wqb, nw4);
    cvt_f32_bf16_x4<<<(nw4 + 255) / 256, 256, 0, stream>>>(Wk, wkb, nw4);
    cvt_f32_bf16_x4<<<(nw4 + 255) / 256, 256, 0, stream>>>(Wv, wvb, nw4);
    cvt_f32_bf16_x4<<<(nw4 + 255) / 256, 256, 0, stream>>>(Wo, wob, nw4);

    dim3 pgrid(BATCH * S_LEN / 64, DMODEL / 64);   // (128, 16)
    // q scaled by 1/sqrt(hd) = 0.125 at projection time
    proj_gemm<<<pgrid, 128, 0, stream>>>(xb, wqb, qb, 0, 0.125f);
    proj_gemm<<<pgrid, 128, 0, stream>>>(xb, wkb, kbf, 0, 1.0f);
    proj_gemm<<<pgrid, 128, 0, stream>>>(xb, wvb, vtb, 1, 1.0f);

    flash_attn<<<dim3(BATCH * NHEADS, S_LEN / 64), 128, 0, stream>>>(qb, kbf, vtb, ab);

    proj_gemm<<<pgrid, 128, 0, stream>>>(ab, wob, d_out, 2, 1.0f);
}
